// SeqCondAttentionV2_39874476376258
// MI455X (gfx1250) — compile-verified, matrix operand order checked by reading, not run on hardware
//
#include <hip/hip_runtime.h>

// ---------------- problem constants ----------------
#define BATCH 2
#define LSEQ  2048
#define DDIM  1024
#define KHEAD 32
#define MROT  4
#define HDIM  32          // D / K
#define AH    4
#define DH    28          // K - AH
#define CKW   4
#define TOTAL 2208        // 2*D + K + K*M
#define NT    8           // L tiles for blocked scan
#define TILE_L (LSEQ / NT)
#define PI3   1.04719755119659774615f

typedef __attribute__((ext_vector_type(16))) __bf16 bf16x16;
typedef __attribute__((ext_vector_type(8)))  __bf16 bf16x8;
typedef __attribute__((ext_vector_type(8)))  float  f32x8;

static __device__ __forceinline__ unsigned short f2bf(float f) {
    unsigned u = __builtin_bit_cast(unsigned, f);
    unsigned r = u + 0x7FFFu + ((u >> 16) & 1u);
    return (unsigned short)(r >> 16);
}

static __device__ __forceinline__ float softplus_f(float x) {
    return (x > 20.f) ? x : log1pf(__expf(x));
}

// ---------------- fp32 -> bf16 convert ----------------
__global__ void cvt_bf16_kernel(const float* __restrict__ src,
                                unsigned short* __restrict__ dst, int n) {
    for (int i = blockIdx.x * blockDim.x + threadIdx.x; i < n;
         i += gridDim.x * blockDim.x)
        dst[i] = f2bf(src[i]);
}

// fp32 (R x C) row-major  ->  bf16 transposed (C x R) row-major
__global__ void cvt_bf16_t_kernel(const float* __restrict__ src,
                                  unsigned short* __restrict__ dst,
                                  int R, int C) {
    for (int i = blockIdx.x * blockDim.x + threadIdx.x; i < R * C;
         i += gridDim.x * blockDim.x) {
        int c = i / R;          // output row
        int r = i % R;          // output col
        dst[i] = f2bf(src[(size_t)r * C + c]);
    }
}

// build [W_re;W_im]^T : dst (32 x 256) bf16, dst[h*256+c]
__global__ void build_wc_t_kernel(const float* __restrict__ W_re,
                                  const float* __restrict__ W_im,
                                  unsigned short* __restrict__ dst) {
    int i = blockIdx.x * blockDim.x + threadIdx.x;  // over 32*256
    if (i >= HDIM * 256) return;
    int h = i / 256, c = i % 256;
    float v = (c < 128) ? W_re[(size_t)c * HDIM + h]
                        : W_im[(size_t)(c - 128) * HDIM + h];
    dst[i] = f2bf(v);
}

// ---------------- generic bf16 WMMA GEMM: C(MxN) = A(MxK) * B(KxN) -------
// A row-major bf16; Bt = B^T row-major bf16 (N x K); C row-major fp32.
// Block: 256 threads (8 waves). Tile: BM=256, BN=32, BK=32.
// Staging: global_load_async_to_lds (ASYNCcnt), double-buffered LDS so the
// next K-tile streams in while WMMAs consume the current one.
#define BM 256
#define BN 32
#define BK 32
#define APITCH 40   // bf16 elems per LDS row (80B, 16B aligned, bank-spread)
#define BPITCH 40

static __device__ __forceinline__ bf16x16 load_frag(const __bf16* rowBase, int kb) {
    bf16x8 lo = *(const bf16x8*)(rowBase + kb);
    bf16x8 hi = *(const bf16x8*)(rowBase + 16 + kb);
    return __builtin_shufflevector(lo, hi, 0,1,2,3,4,5,6,7,8,9,10,11,12,13,14,15);
}

static __device__ __forceinline__ void async_b128(unsigned lds_addr,
                                                  unsigned long long gaddr) {
    asm volatile("global_load_async_to_lds_b128 %0, %1, off"
                 :: "v"(lds_addr), "v"(gaddr) : "memory");
}
static __device__ __forceinline__ void async_b64(unsigned lds_addr,
                                                 unsigned long long gaddr) {
    asm volatile("global_load_async_to_lds_b64 %0, %1, off"
                 :: "v"(lds_addr), "v"(gaddr) : "memory");
}

__global__ __launch_bounds__(256) void wmma_gemm_bf16_kernel(
    const __bf16* __restrict__ A, const __bf16* __restrict__ Bt,
    float* __restrict__ C, int M, int N, int Kd, int lda, int ldb, int ldc)
{
    __shared__ __bf16 As[2][BM * APITCH];
    __shared__ __bf16 Bs[2][BN * BPITCH];

    const int tid  = threadIdx.x;
    const int wave = tid >> 5;
    const int lane = tid & 31;
    const int m0 = blockIdx.y * BM;
    const int n0 = blockIdx.x * BN;

    const int mr = lane & 15;         // row-in-tile for A, col for B
    const int kb = (lane >> 4) * 8;   // K base within fragment half

    // per-thread staging work: 4x b128 for A (256x32), 1x b64 for B (32x32)
    // => exactly 5 async ops per thread per tile (uniform ASYNCcnt).
    auto stage_tile = [&](int k0, int buf) {
        #pragma unroll
        for (int i = 0; i < 4; ++i) {
            int cid = tid + 256 * i;
            int row = cid >> 2;
            int c8  = cid & 3;
            unsigned long long gp = (unsigned long long)(size_t)
                (A + (size_t)(m0 + row) * lda + k0 + c8 * 8);
            unsigned lp = (unsigned)(uintptr_t)&As[buf][row * APITCH + c8 * 8];
            async_b128(lp, gp);
        }
        {
            int row = tid >> 3;       // 0..31  (n within tile)
            int c4  = tid & 7;        // 0..7   (4-elem k chunk)
            unsigned long long gp = (unsigned long long)(size_t)
                (Bt + (size_t)(n0 + row) * ldb + k0 + c4 * 4);
            unsigned lp = (unsigned)(uintptr_t)&Bs[buf][row * BPITCH + c4 * 4];
            async_b64(lp, gp);
        }
    };

    f32x8 acc00 = {}, acc01 = {}, acc10 = {}, acc11 = {};

    const int nsteps = Kd / BK;
    stage_tile(0, 0);

    for (int s = 0; s < nsteps; ++s) {
        const int buf = s & 1;
        if (s + 1 < nsteps) {
            stage_tile((s + 1) * BK, buf ^ 1);
            // current tile's 5 ops done; next tile's 5 may remain in flight
            asm volatile("s_wait_asynccnt 0x5" ::: "memory");
        } else {
            asm volatile("s_wait_asynccnt 0x0" ::: "memory");
        }
        __syncthreads();

        const int mw = wave * 32;
        bf16x16 a0 = load_frag(&As[buf][(mw + mr)      * APITCH], kb);
        bf16x16 a1 = load_frag(&As[buf][(mw + 16 + mr) * APITCH], kb);
        bf16x16 b0 = load_frag(&Bs[buf][ mr            * BPITCH], kb);
        bf16x16 b1 = load_frag(&Bs[buf][(16 + mr)      * BPITCH], kb);

        acc00 = __builtin_amdgcn_wmma_f32_16x16x32_bf16(false, a0, false, b0, (short)0, acc00, false, false);
        acc01 = __builtin_amdgcn_wmma_f32_16x16x32_bf16(false, a0, false, b1, (short)0, acc01, false, false);
        acc10 = __builtin_amdgcn_wmma_f32_16x16x32_bf16(false, a1, false, b0, (short)0, acc10, false, false);
        acc11 = __builtin_amdgcn_wmma_f32_16x16x32_bf16(false, a1, false, b1, (short)0, acc11, false, false);
        __syncthreads();
    }

    // C/D 16x16 layout: lane L -> n = L&15, rows m = (L>>4)*8 + v
    const int mw   = m0 + wave * 32;
    const int half = lane >> 4;
    const int nn   = n0 + (lane & 15);
    #pragma unroll
    for (int v = 0; v < 8; ++v) {
        int mrow = half * 8 + v;
        C[(size_t)(mw + mrow)      * ldc + nn]      = acc00[v];
        C[(size_t)(mw + mrow)      * ldc + nn + 16] = acc01[v];
        C[(size_t)(mw + 16 + mrow) * ldc + nn]      = acc10[v];
        C[(size_t)(mw + 16 + mrow) * ldc + nn + 16] = acc11[v];
    }
}

// ---------------- depthwise causal conv + bias ----------------
__global__ void conv_kernel(const float* __restrict__ z,
                            const float* __restrict__ conv_w,
                            const float* __restrict__ conv_b,
                            float* __restrict__ zc)
{
    size_t i = (size_t)blockIdx.x * blockDim.x + threadIdx.x;
    size_t total = (size_t)BATCH * LSEQ * TOTAL;
    if (i >= total) return;
    int c = (int)(i % TOTAL);
    size_t rl = i / TOTAL;
    int l = (int)(rl % LSEQ);
    size_t b = rl / LSEQ;
    float acc = conv_b[c];
    #pragma unroll
    for (int j = 0; j < CKW; ++j) {
        int ls = l - (CKW - 1) + j;
        if (ls >= 0)
            acc += z[(b * LSEQ + ls) * TOTAL + c] * conv_w[j * TOTAL + c];
    }
    zc[i] = acc;
}

// ---------------- blocked scan, pass A: per-L-tile totals ----------------
__global__ __launch_bounds__(128) void scan_totals_kernel(
    const float* __restrict__ zc, const float* __restrict__ mask,
    const float* __restrict__ theta_base, const float* __restrict__ trs,
    const float* __restrict__ dslope, const float* __restrict__ aslope,
    const float* __restrict__ sscale,
    float* __restrict__ tot_re, float* __restrict__ tot_im,
    float* __restrict__ tot_den)
{
    const int tile = blockIdx.x, k = blockIdx.y, b = blockIdx.z;
    const int c = threadIdx.x;
    const int h = c >> 2, mi = c & 3;
    const float tb = theta_base[k * MROT + mi];
    const float rs = trs[k];
    const float ss = sscale[k];
    const float slope = (k < DH) ? softplus_f(dslope[k]) : softplus_f(aslope[k - DH]);
    float cre = 0.f, cim = 0.f, cden = 0.f;
    const int l0 = tile * TILE_L;
    for (int t = 0; t < TILE_L; ++t) {
        const int l = l0 + t;
        const float* row = zc + ((size_t)b * LSEQ + l) * TOTAL;
        const float mk = mask[b * LSEQ + l];
        float s  = row[2 * DDIM + k] * mk;
        float pv = __expf(fminf(fmaxf(ss * s, -20.f), 20.f)) * mk;
        float dist = (k < DH) ? fmaxf((float)(LSEQ - 1 - l), 0.f) : (float)l;
        float pw = pv * __expf(-slope * dist);
        float xv = row[k * HDIM + h] * mk;
        float tr = row[2 * DDIM + KHEAD + k * MROT + mi];
        float td = tr / (1.f + fabsf(tr));
        float phi = xv * (tb + rs * (td * PI3));
        cre += pw * __cosf(phi);
        cim += pw * __sinf(phi);
        cden += pw;
    }
    size_t base = (((size_t)b * KHEAD + k) * NT + tile) * 128 + c;
    tot_re[base] = cre;
    tot_im[base] = cim;
    if (c == 0) tot_den[((size_t)b * KHEAD + k) * NT + tile] = cden;
}

// ---------------- pass B: exclusive scan of tile totals (in place) -------
__global__ void scan_offsets_kernel(float* __restrict__ tot_re,
                                    float* __restrict__ tot_im,
                                    float* __restrict__ tot_den)
{
    int i = blockIdx.x * blockDim.x + threadIdx.x;
    const int nch = BATCH * KHEAD * 129;
    if (i >= nch) return;
    int bk = i / 129, c = i % 129;
    if (c < 128) {
        float* p = tot_re + (size_t)bk * NT * 128 + c;
        float* q = tot_im + (size_t)bk * NT * 128 + c;
        float r0 = 0.f, r1 = 0.f;
        for (int t = 0; t < NT; ++t) {
            float v0 = p[t * 128]; p[t * 128] = r0; r0 += v0;
            float v1 = q[t * 128]; q[t * 128] = r1; r1 += v1;
        }
    } else {
        float* p = tot_den + (size_t)bk * NT;
        float r = 0.f;
        for (int t = 0; t < NT; ++t) { float v = p[t]; p[t] = r; r += v; }
    }
}

// ---------------- pass C: replay scan, normalize, RMS, pack bf16 --------
__global__ __launch_bounds__(128) void scan_apply_kernel(
    const float* __restrict__ zc, const float* __restrict__ mask,
    const float* __restrict__ theta_base, const float* __restrict__ trs,
    const float* __restrict__ dslope, const float* __restrict__ aslope,
    const float* __restrict__ sscale, const float* __restrict__ norm_scale,
    const float* __restrict__ off_re, const float* __restrict__ off_im,
    const float* __restrict__ off_den, unsigned short* __restrict__ cat)
{
    const int tile = blockIdx.x, k = blockIdx.y, b = blockIdx.z;
    const int c = threadIdx.x;
    const int h = c >> 2, mi = c & 3;
    __shared__ float red[2][4];
    const float tb = theta_base[k * MROT + mi];
    const float rs = trs[k];
    const float ss = sscale[k];
    const float nsr = norm_scale[c];
    const float nsi = norm_scale[128 + c];
    const float slope = (k < DH) ? softplus_f(dslope[k]) : softplus_f(aslope[k - DH]);
    size_t base = (((size_t)b * KHEAD + k) * NT + tile) * 128 + c;
    float cre = off_re[base];
    float cim = off_im[base];
    float cden = off_den[((size_t)b * KHEAD + k) * NT + tile];
    const int l0 = tile * TILE_L;
    for (int t = 0; t < TILE_L; ++t) {
        const int l = l0 + t;
        const float* row = zc + ((size_t)b * LSEQ + l) * TOTAL;
        const float mk = mask[b * LSEQ + l];
        float s  = row[2 * DDIM + k] * mk;
        float pv = __expf(fminf(fmaxf(ss * s, -20.f), 20.f)) * mk;
        float dist = (k < DH) ? fmaxf((float)(LSEQ - 1 - l), 0.f) : (float)l;
        float pw = pv * __expf(-slope * dist);
        float xv = row[k * HDIM + h] * mk;
        float tr = row[2 * DDIM + KHEAD + k * MROT + mi];
        float td = tr / (1.f + fabsf(tr));
        float phi = xv * (tb + rs * (td * PI3));
        cre += pw * __cosf(phi);
        cim += pw * __sinf(phi);
        cden += pw;

        float inv = 1.f / fmaxf(cden, 1e-4f);
        float re = cre * inv, im = cim * inv;

        float v = re * re + im * im;
        #pragma unroll
        for (int d = 16; d > 0; d >>= 1) v += __shfl_xor(v, d, 32);
        if ((threadIdx.x & 31) == 0) red[t & 1][threadIdx.x >> 5] = v;
        __syncthreads();
        float sum = red[t & 1][0] + red[t & 1][1] + red[t & 1][2] + red[t & 1][3];
        float rsq = rsqrtf(sum * (1.f / 256.f) + 1e-5f);

        size_t orow = (((size_t)b * LSEQ + l) * KHEAD + k) * 256;
        cat[orow + c]       = f2bf(re * rsq * nsr);
        cat[orow + 128 + c] = f2bf(im * rsq * nsi);
    }
}

// ---------------- gate epilogue: yg = bf16(yhead * silu(gate)) ----------
__global__ void gate_kernel(const float* __restrict__ yhead,
                            const float* __restrict__ zc,
                            unsigned short* __restrict__ yg)
{
    size_t i = (size_t)blockIdx.x * blockDim.x + threadIdx.x;
    size_t total = (size_t)BATCH * LSEQ * DDIM;
    if (i >= total) return;
    int d = (int)(i % DDIM);
    size_t bl = i / DDIM;
    int h = d & (HDIM - 1), k = d >> 5;
    float y = yhead[(bl * KHEAD + k) * HDIM + h];
    float gin = zc[bl * TOTAL + DDIM + d];
    float g = gin / (1.f + __expf(-gin));
    yg[bl * DDIM + d] = f2bf(y * g);
}

// ---------------- host side ----------------
extern "C" void kernel_launch(void* const* d_in, const int* in_sizes, int n_in,
                              void* d_out, int out_size, void* d_ws, size_t ws_size,
                              hipStream_t stream)
{
    const float* x      = (const float*)d_in[0];
    const float* mask   = (const float*)d_in[1];
    const float* W_in   = (const float*)d_in[2];
    const float* conv_w = (const float*)d_in[3];
    const float* conv_b = (const float*)d_in[4];
    const float* theta_base = (const float*)d_in[5];
    const float* trs    = (const float*)d_in[6];
    const float* dsl    = (const float*)d_in[7];
    const float* asl    = (const float*)d_in[8];
    const float* ssc    = (const float*)d_in[9];
    const float* nsc    = (const float*)d_in[10];
    const float* W_re   = (const float*)d_in[11];
    const float* W_im   = (const float*)d_in[12];
    const float* W_out  = (const float*)d_in[13];

    char* p = (char*)d_ws;
    auto alloc = [&](size_t bytes) {
        void* r = (void*)p;
        p += (bytes + 255) & ~(size_t)255;
        return r;
    };
    const size_t nBL = (size_t)BATCH * LSEQ;                // 4096
    unsigned short* xw     = (unsigned short*)alloc(nBL * DDIM * 2);
    unsigned short* WinT_b = (unsigned short*)alloc((size_t)TOTAL * DDIM * 2); // W_in^T (2208x1024)
    unsigned short* WcT_b  = (unsigned short*)alloc((size_t)HDIM * 256 * 2);   // [W_re;W_im]^T (32x256)
    unsigned short* WoutT_b= (unsigned short*)alloc((size_t)DDIM * DDIM * 2);  // W_out^T
    float* z      = (float*)alloc(nBL * TOTAL * 4);
    float* zc     = (float*)alloc(nBL * TOTAL * 4);
    float* tot_re = (float*)alloc((size_t)BATCH * KHEAD * NT * 128 * 4);
    float* tot_im = (float*)alloc((size_t)BATCH * KHEAD * NT * 128 * 4);
    float* tot_den= (float*)alloc((size_t)BATCH * KHEAD * NT * 4);
    unsigned short* cat = (unsigned short*)alloc(nBL * KHEAD * 256 * 2);
    float* yhead  = (float*)alloc(nBL * KHEAD * HDIM * 4);
    unsigned short* yg = (unsigned short*)alloc(nBL * DDIM * 2);

    // --- fp32 -> bf16 conversions (A matrices plain, B matrices transposed) ---
    cvt_bf16_kernel<<<4096, 256, 0, stream>>>(x, xw, (int)(nBL * DDIM));
    cvt_bf16_t_kernel<<<4096, 256, 0, stream>>>(W_in, WinT_b, DDIM, TOTAL);
    build_wc_t_kernel<<<(HDIM * 256 + 255) / 256, 256, 0, stream>>>(W_re, W_im, WcT_b);
    cvt_bf16_t_kernel<<<4096, 256, 0, stream>>>(W_out, WoutT_b, DDIM, DDIM);

    // --- GEMM1: z(4096 x 2208) = xw(4096 x 1024) @ W_in ---
    wmma_gemm_bf16_kernel<<<dim3(TOTAL / BN, (int)(nBL / BM)), 256, 0, stream>>>(
        (const __bf16*)xw, (const __bf16*)WinT_b, z,
        (int)nBL, TOTAL, DDIM, DDIM, DDIM, TOTAL);

    // --- causal depthwise conv + bias ---
    {
        size_t tot = nBL * TOTAL;
        conv_kernel<<<(unsigned)((tot + 255) / 256), 256, 0, stream>>>(z, conv_w, conv_b, zc);
    }

    // --- blocked scan over L ---
    scan_totals_kernel<<<dim3(NT, KHEAD, BATCH), 128, 0, stream>>>(
        zc, mask, theta_base, trs, dsl, asl, ssc, tot_re, tot_im, tot_den);
    scan_offsets_kernel<<<(BATCH * KHEAD * 129 + 255) / 256, 256, 0, stream>>>(
        tot_re, tot_im, tot_den);
    scan_apply_kernel<<<dim3(NT, KHEAD, BATCH), 128, 0, stream>>>(
        zc, mask, theta_base, trs, dsl, asl, ssc, nsc,
        tot_re, tot_im, tot_den, cat);

    // --- GEMM2: yhead(131072 x 32) = cat(131072 x 256) @ [W_re;W_im](256 x 32) ---
    wmma_gemm_bf16_kernel<<<dim3(1, (int)(nBL * KHEAD / BM)), 256, 0, stream>>>(
        (const __bf16*)cat, (const __bf16*)WcT_b, yhead,
        (int)(nBL * KHEAD), HDIM, 256, 256, 256, HDIM);

    // --- gate epilogue ---
    {
        size_t tot = nBL * DDIM;
        gate_kernel<<<(unsigned)((tot + 255) / 256), 256, 0, stream>>>(yhead, zc, yg);
    }

    // --- GEMM3: out(4096 x 1024) = yg @ W_out ---
    wmma_gemm_bf16_kernel<<<dim3(DDIM / BN, (int)(nBL / BM)), 256, 0, stream>>>(
        (const __bf16*)yg, (const __bf16*)WoutT_b, (float*)d_out,
        (int)nBL, DDIM, DDIM, DDIM, DDIM, DDIM);
}
